// T5Attention_35562329211037
// MI455X (gfx1250) — matrix operand order repaired and checked by reading, not compiled, not run on hardware
//
#include <hip/hip_runtime.h>
#include <hip/hip_bf16.h>

typedef _Float16 half_t;
typedef __attribute__((ext_vector_type(16))) _Float16 v16h;
typedef __attribute__((ext_vector_type(8)))  _Float16 v8h;
typedef __attribute__((ext_vector_type(8)))  float    v8f;

#define Bsz 4
#define Tsz 2048
#define Dsz 512
#define Hh  8
#define DKk 64
#define BT  (Bsz * Tsz)

// ---- WMMA fragment loaders (wave32, 16-bit, per CDNA5 ISA 7.12.2) ----------
// A 16x32 (MxK): lane r=L&15 holds row r; halves 0-7 = K[hi*8 .. hi*8+7],
// halves 8-15 = K[16+hi*8 .. 16+hi*8+7], hi = L>>4.
__device__ __forceinline__ v16h load_a16(const half_t* A, int lda, int m0,
                                         int k0, int lane) {
  int r = lane & 15, hi = lane >> 4;
  const half_t* p = A + (size_t)(m0 + r) * lda + k0 + hi * 8;
  v8h lo = *(const v8h*)p;
  v8h hh = *(const v8h*)(p + 16);
  v16h out;
#pragma unroll
  for (int i = 0; i < 8; ++i) { out[i] = lo[i]; out[8 + i] = hh[i]; }
  return out;
}

// B 32x16 (KxN) where B[k][n] = M[n0+n][k0+k] (rows of row-major M used as
// columns, i.e. the X @ W^T pattern): lane L holds column n=L&15; halves 0-7 =
// K[hi*8 .. hi*8+7], halves 8-15 = K[16+hi*8 .. 16+hi*8+7], hi = L>>4.
__device__ __forceinline__ v16h load_bT(const half_t* M, int ldm, int n0,
                                        int k0, int lane) {
  int n = lane & 15, kb = (lane >> 4) * 8;
  const half_t* p = M + (size_t)(n0 + n) * ldm + k0 + kb;
  v8h lo = *(const v8h*)p;
  v8h hh = *(const v8h*)(p + 16);
  v16h out;
#pragma unroll
  for (int i = 0; i < 8; ++i) { out[i] = lo[i]; out[8 + i] = hh[i]; }
  return out;
}

// ---- f32 -> f16 convert ----------------------------------------------------
__global__ void cvt_f16_kernel(const float* __restrict__ src,
                               half_t* __restrict__ dst, int n) {
  int i = blockIdx.x * blockDim.x + threadIdx.x;
  if (i < n) dst[i] = (half_t)src[i];
}

// ---- QKV projection: q/k/v = x @ W^T ---------------------------------------
// blockIdx.y selects the matrix (0=q,1=k,2=v) -> scalar branch, no exec mask.
// Each wave computes a 16x32 output tile (2 accumulators, shared A fragment).
__global__ void __launch_bounds__(128) qkv_proj_kernel(
    const half_t* __restrict__ xh, const half_t* __restrict__ wq,
    const half_t* __restrict__ wk, const half_t* __restrict__ wv,
    half_t* __restrict__ qh, half_t* __restrict__ kh,
    half_t* __restrict__ vt) {
  int lane  = threadIdx.x & 31;
  int tile  = blockIdx.x * 4 + (threadIdx.x >> 5);
  int which = blockIdx.y;                       // SGPR-uniform
  int m0 = (tile / (Dsz / 32)) * 16;            // row tile in [0, BT)
  int n0 = (tile % (Dsz / 32)) * 32;            // col tile in [0, D)
  const half_t* W = (which == 0) ? wq : (which == 1) ? wk : wv;

  v8f c0 = {}, c1 = {};
#pragma unroll 4
  for (int kk = 0; kk < Dsz; kk += 32) {
    v16h a  = load_a16(xh, Dsz, m0, kk, lane);
    v16h b0 = load_bT(W, Dsz, n0, kk, lane);
    v16h b1 = load_bT(W, Dsz, n0 + 16, kk, lane);
    c0 = __builtin_amdgcn_wmma_f32_16x16x32_f16(false, a, false, b0, (short)0,
                                                c0, false, false);
    c1 = __builtin_amdgcn_wmma_f32_16x16x32_f16(false, a, false, b1, (short)0,
                                                c1, false, false);
  }
  int hi = lane >> 4, n = lane & 15;
  if (which == 0) {
#pragma unroll
    for (int j = 0; j < 8; ++j) {
      int M = m0 + j + 8 * hi, bb = M >> 11, t = M & (Tsz - 1);
      int c = n0 + n;
      qh[((size_t)(bb * Hh + (c >> 6)) * Tsz + t) * DKk + (c & 63)] =
          (half_t)c0[j];
      c = n0 + 16 + n;
      qh[((size_t)(bb * Hh + (c >> 6)) * Tsz + t) * DKk + (c & 63)] =
          (half_t)c1[j];
    }
  } else if (which == 1) {
#pragma unroll
    for (int j = 0; j < 8; ++j) {
      int M = m0 + j + 8 * hi, bb = M >> 11, t = M & (Tsz - 1);
      int c = n0 + n;
      kh[((size_t)(bb * Hh + (c >> 6)) * Tsz + t) * DKk + (c & 63)] =
          (half_t)c0[j];
      c = n0 + 16 + n;
      kh[((size_t)(bb * Hh + (c >> 6)) * Tsz + t) * DKk + (c & 63)] =
          (half_t)c1[j];
    }
  } else {  // v stored transposed [B,H,DK,T] for contiguous PV B-operands
#pragma unroll
    for (int j = 0; j < 8; ++j) {
      int M = m0 + j + 8 * hi, bb = M >> 11, t = M & (Tsz - 1);
      int c = n0 + n;
      vt[((size_t)(bb * Hh + (c >> 6)) * DKk + (c & 63)) * Tsz + t] =
          (half_t)c0[j];
      c = n0 + 16 + n;
      vt[((size_t)(bb * Hh + (c >> 6)) * DKk + (c & 63)) * Tsz + t] =
          (half_t)c1[j];
    }
  }
}

// ---- Flash attention: one wave = one (b,h,16-row Q tile) -------------------
__global__ void __launch_bounds__(128) flash_attn_kernel(
    const half_t* __restrict__ qh, const half_t* __restrict__ kh,
    const half_t* __restrict__ vt, const float* __restrict__ bias,
    const float* __restrict__ mask, half_t* __restrict__ yh) {
  __shared__ __align__(64) half_t plds[4][16 * 32];
  int lane = threadIdx.x & 31;
  int wid  = threadIdx.x >> 5;
  int tile = blockIdx.x * 4 + wid;
  int mt = tile & (Tsz / 16 - 1);
  int h  = (tile / (Tsz / 16)) & (Hh - 1);
  int b  = tile / ((Tsz / 16) * Hh);
  int m0 = mt * 16;
  const half_t* qhead = qh + (size_t)(b * Hh + h) * Tsz * DKk;
  const half_t* khead = kh + (size_t)(b * Hh + h) * Tsz * DKk;
  const half_t* vhead = vt + (size_t)(b * Hh + h) * DKk * Tsz;
  const float*  bh    = bias + (size_t)h * Tsz * Tsz;
  const float*  mb    = mask + (size_t)b * Tsz;
  int hi = lane >> 4, n = lane & 15;

  // Q fragments stay resident across the whole S loop (16 VGPRs).
  v16h aq0 = load_a16(qhead, DKk, m0, 0, lane);
  v16h aq1 = load_a16(qhead, DKk, m0, 32, lane);

  float mi[8], li[8];
  v8f y0 = {}, y1 = {}, y2 = {}, y3 = {};
#pragma unroll
  for (int j = 0; j < 8; ++j) { mi[j] = -3.0e38f; li[j] = 0.0f; }

  for (int s0 = 0; s0 < Tsz; s0 += 32) {
    // Prefetch next iteration's position_bias tile (dominant data stream).
    const float* bpn = bh + (size_t)(m0 + 8 * hi) * Tsz + s0 + 32 + n;
#pragma unroll
    for (int j = 0; j < 8; ++j)
      __builtin_prefetch(bpn + (size_t)j * Tsz, 0, 1);

    // S tile 16x32 = two 16x16 C tiles, each = Q(16x64) @ K[s..s+16)^T
    v8f c0 = {}, c1 = {};
    v16h bk = load_bT(khead, DKk, s0, 0, lane);
    c0 = __builtin_amdgcn_wmma_f32_16x16x32_f16(false, aq0, false, bk, (short)0,
                                                c0, false, false);
    bk = load_bT(khead, DKk, s0, 32, lane);
    c0 = __builtin_amdgcn_wmma_f32_16x16x32_f16(false, aq1, false, bk, (short)0,
                                                c0, false, false);
    bk = load_bT(khead, DKk, s0 + 16, 0, lane);
    c1 = __builtin_amdgcn_wmma_f32_16x16x32_f16(false, aq0, false, bk, (short)0,
                                                c1, false, false);
    bk = load_bT(khead, DKk, s0 + 16, 32, lane);
    c1 = __builtin_amdgcn_wmma_f32_16x16x32_f16(false, aq1, false, bk, (short)0,
                                                c1, false, false);

    // additive position bias + mask (T5: no 1/sqrt(dk) scale)
    float mk0 = mb[s0 + n];
    float mk1 = mb[s0 + 16 + n];
    const float* bp = bh + (size_t)(m0 + 8 * hi) * Tsz + s0 + n;
    float p0[8], p1[8];
#pragma unroll
    for (int j = 0; j < 8; ++j) {
      p0[j] = c0[j] + bp[(size_t)j * Tsz] + mk0;
      p1[j] = c1[j] + bp[(size_t)j * Tsz + 16] + mk1;
    }

    // online softmax: row stats per C-VGPR, reduced across each 16-lane half
    // (xor 1,2,4,8 stays inside the half-wave owning rows j / j+8).
#pragma unroll
    for (int j = 0; j < 8; ++j) {
      float r = fmaxf(p0[j], p1[j]);
      r = fmaxf(r, __shfl_xor(r, 1, 32));
      r = fmaxf(r, __shfl_xor(r, 2, 32));
      r = fmaxf(r, __shfl_xor(r, 4, 32));
      r = fmaxf(r, __shfl_xor(r, 8, 32));
      float mnew  = fmaxf(mi[j], r);
      float alpha = __expf(mi[j] - mnew);
      mi[j] = mnew;
      p0[j] = __expf(p0[j] - mnew);
      p1[j] = __expf(p1[j] - mnew);
      float s = p0[j] + p1[j];
      s += __shfl_xor(s, 1, 32);
      s += __shfl_xor(s, 2, 32);
      s += __shfl_xor(s, 4, 32);
      s += __shfl_xor(s, 8, 32);
      li[j] = li[j] * alpha + s;
      y0[j] *= alpha; y1[j] *= alpha; y2[j] *= alpha; y3[j] *= alpha;
    }

    // C-layout P -> LDS -> A-layout P (per-wave private 1KB tile; LDS ops
    // from one wave are in-order, no barrier needed).
    half_t* pw = plds[wid];
#pragma unroll
    for (int j = 0; j < 8; ++j) {
      pw[(j + 8 * hi) * 32 + n]      = (half_t)p0[j];
      pw[(j + 8 * hi) * 32 + 16 + n] = (half_t)p1[j];
    }
    v8h plo = *(const v8h*)&pw[(lane & 15) * 32 + hi * 8];
    v8h phh = *(const v8h*)&pw[(lane & 15) * 32 + 16 + hi * 8];
    v16h ap;
#pragma unroll
    for (int i = 0; i < 8; ++i) { ap[i] = plo[i]; ap[8 + i] = phh[i]; }

    // Y(16x64) += P(16x32) @ V(32x64); V transposed so B loads contiguous.
    v16h bv = load_bT(vhead, Tsz, 0, s0, lane);
    y0 = __builtin_amdgcn_wmma_f32_16x16x32_f16(false, ap, false, bv, (short)0,
                                                y0, false, false);
    bv = load_bT(vhead, Tsz, 16, s0, lane);
    y1 = __builtin_amdgcn_wmma_f32_16x16x32_f16(false, ap, false, bv, (short)0,
                                                y1, false, false);
    bv = load_bT(vhead, Tsz, 32, s0, lane);
    y2 = __builtin_amdgcn_wmma_f32_16x16x32_f16(false, ap, false, bv, (short)0,
                                                y2, false, false);
    bv = load_bT(vhead, Tsz, 48, s0, lane);
    y3 = __builtin_amdgcn_wmma_f32_16x16x32_f16(false, ap, false, bv, (short)0,
                                                y3, false, false);
  }

  // normalize and write y as [B,T,H*DK] f16 (input of the Wo GEMM)
#pragma unroll
  for (int j = 0; j < 8; ++j) {
    float inv = 1.0f / li[j];
    int t = m0 + 8 * hi + j;
    size_t row = ((size_t)b * Tsz + t) * Dsz + h * DKk + n;
    yh[row + 0]  = (half_t)(y0[j] * inv);
    yh[row + 16] = (half_t)(y1[j] * inv);
    yh[row + 32] = (half_t)(y2[j] * inv);
    yh[row + 48] = (half_t)(y3[j] * inv);
  }
}

// ---- Output projection: out = y @ Wo^T (f32 out), 16x32 tiles --------------
__global__ void __launch_bounds__(128) out_proj_kernel(
    const half_t* __restrict__ yh, const half_t* __restrict__ wo,
    float* __restrict__ out) {
  int lane = threadIdx.x & 31;
  int tile = blockIdx.x * 4 + (threadIdx.x >> 5);
  int m0 = (tile / (Dsz / 32)) * 16;
  int n0 = (tile % (Dsz / 32)) * 32;
  v8f c0 = {}, c1 = {};
#pragma unroll 4
  for (int kk = 0; kk < Dsz; kk += 32) {
    v16h a  = load_a16(yh, Dsz, m0, kk, lane);
    v16h b0 = load_bT(wo, Dsz, n0, kk, lane);
    v16h b1 = load_bT(wo, Dsz, n0 + 16, kk, lane);
    c0 = __builtin_amdgcn_wmma_f32_16x16x32_f16(false, a, false, b0, (short)0,
                                                c0, false, false);
    c1 = __builtin_amdgcn_wmma_f32_16x16x32_f16(false, a, false, b1, (short)0,
                                                c1, false, false);
  }
  int hi = lane >> 4, n = lane & 15;
#pragma unroll
  for (int j = 0; j < 8; ++j) {
    size_t row = (size_t)(m0 + j + 8 * hi) * Dsz;
    out[row + n0 + n]      = c0[j];
    out[row + n0 + 16 + n] = c1[j];
  }
}

extern "C" void kernel_launch(void* const* d_in, const int* in_sizes, int n_in,
                              void* d_out, int out_size, void* d_ws,
                              size_t ws_size, hipStream_t stream) {
  (void)in_sizes; (void)n_in; (void)out_size; (void)ws_size;
  const float* x    = (const float*)d_in[0];
  const float* mask = (const float*)d_in[1];
  const float* bias = (const float*)d_in[2];
  const float* Wq   = (const float*)d_in[3];
  const float* Wk   = (const float*)d_in[4];
  const float* Wv   = (const float*)d_in[5];
  const float* Wo   = (const float*)d_in[6];
  float* out = (float*)d_out;

  char* base = (char*)d_ws;
  size_t off = 0;
  auto alloc_h = [&](size_t nhalves) {
    half_t* p = (half_t*)(base + off);
    off = (off + nhalves * sizeof(half_t) + 255) & ~(size_t)255;
    return p;
  };
  half_t* xh  = alloc_h((size_t)BT * Dsz);
  half_t* wqh = alloc_h((size_t)Dsz * Dsz);
  half_t* wkh = alloc_h((size_t)Dsz * Dsz);
  half_t* wvh = alloc_h((size_t)Dsz * Dsz);
  half_t* woh = alloc_h((size_t)Dsz * Dsz);
  half_t* qh  = alloc_h((size_t)Bsz * Hh * Tsz * DKk);
  half_t* kh  = alloc_h((size_t)Bsz * Hh * Tsz * DKk);
  half_t* vt  = alloc_h((size_t)Bsz * Hh * DKk * Tsz);
  half_t* yh  = alloc_h((size_t)BT * Dsz);

  cvt_f16_kernel<<<(BT * Dsz) / 256, 256, 0, stream>>>(x, xh, BT * Dsz);
  cvt_f16_kernel<<<(Dsz * Dsz) / 256, 256, 0, stream>>>(Wq, wqh, Dsz * Dsz);
  cvt_f16_kernel<<<(Dsz * Dsz) / 256, 256, 0, stream>>>(Wk, wkh, Dsz * Dsz);
  cvt_f16_kernel<<<(Dsz * Dsz) / 256, 256, 0, stream>>>(Wv, wvh, Dsz * Dsz);
  cvt_f16_kernel<<<(Dsz * Dsz) / 256, 256, 0, stream>>>(Wo, woh, Dsz * Dsz);

  // (BT/16)*(D/32) 16x32 tiles per matrix, 4 waves/block, grid.y = matrix
  dim3 qkv_grid((BT / 16) * (Dsz / 32) / 4, 3);
  qkv_proj_kernel<<<qkv_grid, 128, 0, stream>>>(xh, wqh, wkh, wvh, qh, kh, vt);

  // B*H*(T/16) Q tiles, 4 waves per block
  flash_attn_kernel<<<Bsz * Hh * (Tsz / 16) / 4, 128, 0, stream>>>(
      qh, kh, vt, bias, mask, yh);

  out_proj_kernel<<<(BT / 16) * (Dsz / 32) / 4, 128, 0, stream>>>(yh, woh, out);
}
